// VSSM_Official_23699629539957
// MI455X (gfx1250) — matrix operand order repaired and checked by reading, not compiled, not run on hardware
//
#include <hip/hip_runtime.h>
#include <cstdint>

#define TPB 256

typedef __attribute__((ext_vector_type(16))) __bf16 bf16x16;
typedef __attribute__((ext_vector_type(8)))  float  f32x8;

// ---- WMMA fragment index maps (CDNA5 ISA 7.12.2, 16-bit A 16x32 / B 32x16) ----
__device__ __forceinline__ int amapk(int e, int hi){
  int v = e >> 1, h = e & 1;
  int k = (v < 4) ? (2*v + h) : (16 + 2*(v-4) + h);
  return k + 8*hi;
}
__device__ __forceinline__ int bmapk(int e, int hi){
  int v = e >> 1, h = e & 1;
  int k = (v < 4) ? (2*v + h) : (16 + 2*(v-4) + h);
  return k + 8*hi;
}
__device__ __forceinline__ f32x8 wmma_bf16(bf16x16 a, bf16x16 b, f32x8 c){
  return __builtin_amdgcn_wmma_f32_16x16x32_bf16(false, a, false, b, (short)0, c, false, false);
}

// ---- CDNA5 async global->LDS copy (ASYNCcnt-tracked), ISA 15.18.3 op 98 ----
__device__ __forceinline__ void async_ld16(uint32_t lds_byte_addr, const void* gaddr){
  asm volatile("global_load_async_to_lds_b128 %0, %1, off"
               :: "v"(lds_byte_addr), "v"((unsigned long long)(uintptr_t)gaddr)
               : "memory");
}
__device__ __forceinline__ void wait_async(){
  asm volatile("s_wait_asynccnt 0x0" ::: "memory");
}

__device__ __forceinline__ float siluf(float x){ return x / (1.f + expf(-x)); }
__device__ __forceinline__ float softplusf(float x){ return (x > 20.f) ? x : log1pf(expf(x)); }
__device__ __forceinline__ float geluf(float x){ return 0.5f * x * (1.f + erff(x * 0.7071067811865475f)); }

// ======================= generic GEMM: C = act(A @ W^T + bias) + res =======================
// A: MxK row-major f32, W: NxK row-major f32 (computes A@W.T), out MxN f32.
// Block tile 128x64; 512 threads = 16 waves; each wave owns TWO 16x16 M-tiles sharing one
// B fragment -> 2 WMMAs per k-step per wave. Interior tiles staged with
// global_load_async_to_lds_b128 (3 x 16B per thread per k-step).
__global__ __launch_bounds__(512) void k_gemm(const float* __restrict__ A,
    const float* __restrict__ W, const float* __restrict__ bias,
    const float* __restrict__ res, float* __restrict__ C,
    long M, int N, int K, int act)
{
  __shared__ float sA[128][36];   // 144B row stride -> 16B aligned segments
  __shared__ float sW[64][36];
  long m0 = (long)blockIdx.y * 128;
  int  n0 = blockIdx.x * 64;
  int tid = threadIdx.x, lane = tid & 31, wid = tid >> 5;
  int tr = wid >> 2, tc = wid & 3, ml = lane & 15, hi = lane >> 4;

  bool fullMN = (m0 + 128 <= M) && (n0 + 64 <= N);   // block-uniform
  int rowL = tid >> 3;                                // 0..63
  int seg  = tid & 7;                                 // 8 x 16B segments per 32-float slab
  uint32_t ldsA0 = (uint32_t)(uintptr_t)&sA[rowL][seg*4];
  uint32_t ldsA1 = (uint32_t)(uintptr_t)&sA[rowL + 64][seg*4];
  uint32_t ldsW  = (uint32_t)(uintptr_t)&sW[rowL][seg*4];
  const float* gA0 = A + (m0 + rowL)*(long)K + seg*4;
  const float* gA1 = A + (m0 + 64 + rowL)*(long)K + seg*4;
  const float* gW  = W + ((long)(n0 + rowL))*K + seg*4;

  f32x8 acc0 = {}, acc1 = {};
  for (int k0 = 0; k0 < K; k0 += 32) {
    if (fullMN && (k0 + 32 <= K)) {
      async_ld16(ldsA0, gA0 + k0);
      async_ld16(ldsA1, gA1 + k0);
      async_ld16(ldsW,  gW  + k0);
      if (k0 + 64 <= K) {                 // speculative prefetch of next slab
        __builtin_prefetch(gA0 + k0 + 32, 0, 1);
        __builtin_prefetch(gA1 + k0 + 32, 0, 1);
        __builtin_prefetch(gW  + k0 + 32, 0, 1);
      }
      wait_async();
    } else {
      for (int e = tid; e < 4096; e += 512) {
        int i = e >> 5, kk = e & 31;
        int k = k0 + kk;
        long ra = m0 + i;
        sA[i][kk] = (ra < M && k < K) ? A[ra*(long)K + k] : 0.f;
      }
      for (int e = tid; e < 2048; e += 512) {
        int i = e >> 5, kk = e & 31;
        int k = k0 + kk;
        int rn = n0 + i;
        sW[i][kk] = (rn < N && k < K) ? W[(long)rn*K + k] : 0.f;
      }
    }
    __syncthreads();
    bf16x16 a0, a1, b;
#pragma unroll
    for (int e = 0; e < 16; ++e) {
      int ka = amapk(e,hi), kb = bmapk(e,hi);
      a0[e] = (__bf16) sA[tr*32 + ml][ka];
      a1[e] = (__bf16) sA[tr*32 + 16 + ml][ka];
      b[e]  = (__bf16) sW[tc*16 + ml][kb];
    }
    acc0 = wmma_bf16(a0, b, acc0);
    acc1 = wmma_bf16(a1, b, acc1);
    __syncthreads();
  }
#pragma unroll
  for (int r = 0; r < 8; ++r) {
    int col = n0 + tc*16 + ml;
    long row0 = m0 + tr*32 + r + 8*hi;
    long row1 = row0 + 16;
    if (row0 < M && col < N) {
      float v = acc0[r];
      if (bias) v += bias[col];
      if (act == 1) v = geluf(v);
      if (res) v += res[row0*(long)N + col];
      C[row0*(long)N + col] = v;
    }
    if (row1 < M && col < N) {
      float v = acc1[r];
      if (bias) v += bias[col];
      if (act == 1) v = geluf(v);
      if (res) v += res[row1*(long)N + col];
      C[row1*(long)N + col] = v;
    }
  }
}

// ======================= patch embed (4x4 stride-4 conv, NCHW -> NHWC) =======================
__global__ void k_patch(const float* __restrict__ x, const float* __restrict__ w,
                        const float* __restrict__ b, float* __restrict__ o, int Bn){
  int t = blockIdx.x*TPB + threadIdx.x;
  int total = Bn*64*64*96;
  if (t >= total) return;
  int oc = t % 96; int r = t / 96;
  int ow = r % 64; r /= 64; int oh = r % 64; int bb = r / 64;
  float acc = b[oc];
  for (int ic = 0; ic < 3; ++ic)
    for (int kh = 0; kh < 4; ++kh)
      for (int kw = 0; kw < 4; ++kw)
        acc += x[(((long)(bb*3 + ic)*256 + (oh*4+kh))*256 + (ow*4+kw))]
             * w[((oc*3 + ic)*4 + kh)*4 + kw];
  o[t] = acc;
}

// ======================= LayerNorm over last dim =======================
__global__ __launch_bounds__(256) void k_ln(const float* __restrict__ x,
    const float* __restrict__ w, const float* __restrict__ b,
    float* __restrict__ o, long rows, int C)
{
  long row = blockIdx.x;
  const float* xr = x + row*(long)C;
  float* orow = o + row*(long)C;
  __shared__ float rs[256], rs2[256];
  int tid = threadIdx.x;
  float s = 0.f, s2 = 0.f;
  for (int i = tid; i < C; i += 256) { float v = xr[i]; s += v; s2 += v*v; }
  rs[tid] = s; rs2[tid] = s2; __syncthreads();
  for (int off = 128; off > 0; off >>= 1) {
    if (tid < off) { rs[tid] += rs[tid+off]; rs2[tid] += rs2[tid+off]; }
    __syncthreads();
  }
  float mu = rs[0] / C;
  float var = rs2[0] / C - mu*mu;
  float inv = rsqrtf(var + 1e-5f);
  for (int i = tid; i < C; i += 256)
    orow[i] = (xr[i] - mu) * inv * w[i] + b[i];
}

// ======================= cross-scan gather: (B,S,S,C) -> (4B, L, C) =======================
__global__ void k_gather(const float* __restrict__ x, float* __restrict__ seqs,
                         int Bn, int S, int C){
  long t = blockIdx.x*(long)TPB + threadIdx.x;
  long L = (long)S*S;
  long total = 4L*Bn*L*C;
  if (t >= total) return;
  int ch = t % C; long r = t / C;
  int l = (int)(r % L); long sb = r / L;
  int b = (int)(sb % Bn); int dir = (int)(sb / Bn);
  int hh, ww;
  if (dir == 0)      { hh = l/S; ww = l%S; }
  else if (dir == 1) { hh = l/S; ww = S-1-(l%S); }
  else if (dir == 2) { ww = l/S; hh = l%S; }
  else               { ww = l/S; hh = S-1-(l%S); }
  seqs[t] = x[(((long)b*S + hh)*S + ww)*C + ch];
}

// ======================= combine 4 directions + residual (in place) =======================
__global__ void k_combine(float* __restrict__ x, const float* __restrict__ m,
                          int Bn, int S, int C){
  long t = blockIdx.x*(long)TPB + threadIdx.x;
  long total = (long)Bn*S*S*C;
  if (t >= total) return;
  int ch = t % C; long r = t / C;
  int ww = (int)(r % S); r /= S; int hh = (int)(r % S); int b = (int)(r / S);
  long L = (long)S*S;
  long l0 = (long)hh*S + ww;
  long l1 = (long)hh*S + (S-1-ww);
  long l2 = (long)ww*S + hh;
  long l3 = (long)ww*S + (S-1-hh);
  float v = m[(((long)(0*Bn + b))*L + l0)*C + ch]
          + m[(((long)(1*Bn + b))*L + l1)*C + ch]
          + m[(((long)(2*Bn + b))*L + l2)*C + ch]
          + m[(((long)(3*Bn + b))*L + l3)*C + ch];
  x[t] += 0.25f * v;
}

// ======================= dt = softplus(zx[..., di+Dc+h] + dt_bias) =======================
__global__ void k_dt(const float* __restrict__ zx, const float* __restrict__ dt_bias,
                     float* __restrict__ dt, long rows, int dip, int di, int Dc, int H){
  long t = blockIdx.x*(long)TPB + threadIdx.x;
  if (t >= rows*(long)H) return;
  int h = (int)(t % H); long r = t / H;
  dt[t] = softplusf(zx[r*(long)dip + di + Dc + h] + dt_bias[h]);
}

// ======================= depthwise causal conv (dconv=4) + bias + silu =======================
__global__ void k_conv(const float* __restrict__ zx, const float* __restrict__ cw,
                       const float* __restrict__ cb, float* __restrict__ xbc,
                       int nseq, int L, int Dc, int dip, int di){
  long t = blockIdx.x*(long)TPB + threadIdx.x;
  long total = (long)nseq*L*Dc;
  if (t >= total) return;
  int ch = (int)(t % Dc); long r = t / Dc;
  int l = (int)(r % L); int sq = (int)(r / L);
  float acc = cb[ch];
#pragma unroll
  for (int k = 0; k < 4; ++k) {
    int ll = l - 3 + k;
    if (ll >= 0) acc += cw[ch*4 + k] * zx[((long)sq*L + ll)*dip + di + ch];
  }
  xbc[t] = siluf(acc);
}

// ======================= SSD part 1: per (seq, chunk, head) =======================
// G = Cc*Bc^T (masked/scaled by Lmat), Y_diag = G'*X, states = X^T*(B*decay).
// Q=P=N=64 fixed; 512 threads = 16 waves, each owns one 16x16 WMMA tile.
__global__ __launch_bounds__(512) void k_ssd1(const float* __restrict__ xbc,
    const float* __restrict__ dt, const float* __restrict__ A_log,
    float* __restrict__ Y, float* __restrict__ states, float* __restrict__ cumA_g,
    int L, int Cn, int H, int Dc, int di)
{
  int idx = blockIdx.x;
  int h = idx % H; int c = (idx / H) % Cn; int bb = idx / (H*Cn);
  int l0 = c*64;
  __shared__ float sB[64][65], sX[64][65], sC[64][65];
  __shared__ float cumA[64];
  int tid = threadIdx.x;
  for (int e = tid; e < 4096; e += 512) {
    int l = e >> 6, j = e & 63;
    long row = (long)bb*L + l0 + l;
    const float* xr = xbc + row*(long)Dc;
    sB[l][j] = xr[di + j];
    sC[l][j] = xr[di + 64 + j];
    sX[l][j] = xr[h*64 + j] * dt[row*(long)H + h];
  }
  if (tid == 0) {
    float Ah = -expf(A_log[h]);
    float s = 0.f;
    for (int l = 0; l < 64; ++l) {
      long row = (long)bb*L + l0 + l;
      s += dt[row*(long)H + h] * Ah;
      cumA[l] = s;
    }
  }
  __syncthreads();
  if (tid < 64) cumA_g[((long)(bb*Cn + c)*H + h)*64 + tid] = cumA[tid];

  int lane = tid & 31, wid = tid >> 5;
  int tr = wid >> 2, tc = wid & 3, ml = lane & 15, hi = lane >> 4;

  // GEMM1: G[l,s] = sum_n C[l,n] * B[s,n]
  f32x8 acc = {};
  for (int k0 = 0; k0 < 64; k0 += 32) {
    bf16x16 a, b;
#pragma unroll
    for (int e = 0; e < 16; ++e) {
      a[e] = (__bf16) sC[tr*16 + ml][k0 + amapk(e,hi)];
      b[e] = (__bf16) sB[tc*16 + ml][k0 + bmapk(e,hi)];
    }
    acc = wmma_bf16(a, b, acc);
  }
  __syncthreads();
  // apply causal decay mask (Lmat) and stash G' in sC
#pragma unroll
  for (int r = 0; r < 8; ++r) {
    int l = tr*16 + r + 8*hi;
    int s = tc*16 + ml;
    sC[l][s] = (s <= l) ? acc[r] * expf(cumA[l] - cumA[s]) : 0.f;
  }
  __syncthreads();

  // GEMM2: Y_diag[l,p] = sum_s G'[l,s] * X[s,p]
  f32x8 acc2 = {};
  for (int k0 = 0; k0 < 64; k0 += 32) {
    bf16x16 a, b;
#pragma unroll
    for (int e = 0; e < 16; ++e) {
      a[e] = (__bf16) sC[tr*16 + ml][k0 + amapk(e,hi)];
      b[e] = (__bf16) sX[k0 + bmapk(e,hi)][tc*16 + ml];
    }
    acc2 = wmma_bf16(a, b, acc2);
  }
#pragma unroll
  for (int r = 0; r < 8; ++r) {
    int l = tr*16 + r + 8*hi;
    int p = tc*16 + ml;
    long row = (long)bb*L + l0 + l;
    Y[row*(long)di + h*64 + p] = acc2[r];
  }

  // GEMM3: states[p,n] = sum_l X[l,p] * decay[l] * B[l,n]
  f32x8 acc3 = {};
  float Aend = cumA[63];
  for (int k0 = 0; k0 < 64; k0 += 32) {
    bf16x16 a, b;
#pragma unroll
    for (int e = 0; e < 16; ++e) {
      int ka = k0 + amapk(e,hi);
      int kb = k0 + bmapk(e,hi);
      a[e] = (__bf16) sX[ka][tr*16 + ml];
      b[e] = (__bf16) (sB[kb][tc*16 + ml] * expf(Aend - cumA[kb]));
    }
    acc3 = wmma_bf16(a, b, acc3);
  }
#pragma unroll
  for (int r = 0; r < 8; ++r) {
    int p = tr*16 + r + 8*hi;
    int n = tc*16 + ml;
    states[(((long)(bb*Cn + c)*H + h) << 12) + p*64 + n] = acc3[r];
  }
}

// ======================= inter-chunk state recurrence (in place) =======================
__global__ void k_scan(float* __restrict__ states, const float* __restrict__ cumA_g,
                       int nseq, int Cn, int H){
  long t = blockIdx.x*(long)TPB + threadIdx.x;
  long total = ((long)nseq*H) << 12;
  if (t >= total) return;
  int pn = (int)(t & 4095);
  int h  = (int)((t >> 12) % H);
  int bb = (int)(t / ((long)H << 12));
  float s = 0.f;
  for (int z = 0; z < Cn; ++z) {
    long base = (((long)(bb*Cn + z)*H + h) << 12) + pn;
    float tmp = states[base];
    states[base] = s;  // state entering chunk z
    float Aend = cumA_g[((long)(bb*Cn + z)*H + h)*64 + 63];
    s = s*expf(Aend) + tmp;
  }
}

// ======================= SSD part 2: Y += exp(cumA)*C@state^T + D*xh =======================
__global__ __launch_bounds__(512) void k_ssd2(const float* __restrict__ xbc,
    const float* __restrict__ states, const float* __restrict__ cumA_g,
    const float* __restrict__ Dp, float* __restrict__ Y,
    int L, int Cn, int H, int Dc, int di)
{
  int idx = blockIdx.x;
  int h = idx % H; int c = (idx / H) % Cn; int bb = idx / (H*Cn);
  int l0 = c*64;
  __shared__ float sC[64][65], sS[64][65];
  __shared__ float cumA[64];
  int tid = threadIdx.x;
  long sbase = ((long)(bb*Cn + c)*H + h) << 12;
  for (int e = tid; e < 4096; e += 512) {
    int l = e >> 6, j = e & 63;
    long row = (long)bb*L + l0 + l;
    sC[l][j] = xbc[row*(long)Dc + di + 64 + j];
    sS[l][j] = states[sbase + ((long)l << 6) + j];   // sS[p][n]
  }
  if (tid < 64) cumA[tid] = cumA_g[((long)(bb*Cn + c)*H + h)*64 + tid];
  __syncthreads();

  int lane = tid & 31, wid = tid >> 5;
  int tr = wid >> 2, tc = wid & 3, ml = lane & 15, hi = lane >> 4;
  f32x8 acc = {};
  for (int k0 = 0; k0 < 64; k0 += 32) {
    bf16x16 a, b;
#pragma unroll
    for (int e = 0; e < 16; ++e) {
      a[e] = (__bf16) sC[tr*16 + ml][k0 + amapk(e,hi)];
      b[e] = (__bf16) sS[tc*16 + ml][k0 + bmapk(e,hi)];
    }
    acc = wmma_bf16(a, b, acc);
  }
  float Dh = Dp[h];
#pragma unroll
  for (int r = 0; r < 8; ++r) {
    int l = tr*16 + r + 8*hi;
    int p = tc*16 + ml;
    long row = (long)bb*L + l0 + l;
    float xh = xbc[row*(long)Dc + h*64 + p];
    Y[row*(long)di + h*64 + p] += expf(cumA[l]) * acc[r] + Dh * xh;
  }
}

// ======================= gate with silu(z) then RMS-norm (in place on Y) =======================
__global__ __launch_bounds__(256) void k_gate_rms(float* __restrict__ Y,
    const float* __restrict__ zx, const float* __restrict__ nw,
    long rows, int di, int dip)
{
  long row = blockIdx.x;
  const float* zr = zx + row*(long)dip;
  float* yr = Y + row*(long)di;
  __shared__ float red[256];
  int tid = threadIdx.x;
  float ss = 0.f;
  for (int i = tid; i < di; i += 256) {
    float z = zr[i];
    float y = yr[i] * siluf(z);
    yr[i] = y;
    ss += y*y;
  }
  red[tid] = ss; __syncthreads();
  for (int off = 128; off > 0; off >>= 1) {
    if (tid < off) red[tid] += red[tid+off];
    __syncthreads();
  }
  float scale = rsqrtf(red[0]/di + 1e-5f);
  for (int i = tid; i < di; i += 256)
    yr[i] = yr[i] * scale * nw[i];
}

// ======================= patch merging gather: (B,S,S,C) -> (B,S/2,S/2,4C) =======================
__global__ void k_merge_gather(const float* __restrict__ x, float* __restrict__ o,
                               int Bn, int S, int C){
  long t = blockIdx.x*(long)TPB + threadIdx.x;
  int S2 = S/2;
  long total = (long)Bn*S2*S2*4*C;
  if (t >= total) return;
  int cc = (int)(t % (4*C)); long r = t / (4*C);
  int j = (int)(r % S2); r /= S2; int i = (int)(r % S2); int b = (int)(r / S2);
  int q = cc / C, ch = cc % C;
  int dr = (q == 1 || q == 3) ? 1 : 0;
  int ds = (q >= 2) ? 1 : 0;
  o[t] = x[(((long)b*S + (2*i + dr))*S + (2*j + ds))*C + ch];
}

// ======================= mean pool over spatial =======================
__global__ void k_meanpool(const float* __restrict__ x, float* __restrict__ o,
                           int Bn, int S, int C){
  int t = blockIdx.x*TPB + threadIdx.x;
  if (t >= Bn*C) return;
  int ch = t % C; int b = t / C;
  float s = 0.f;
  for (int i = 0; i < S*S; ++i) s += x[((long)b*S*S + i)*C + ch];
  o[t] = s / (float)(S*S);
}

// ============================================================================================
extern "C" void kernel_launch(void* const* d_in, const int* in_sizes, int n_in,
                              void* d_out, int out_size, void* d_ws, size_t ws_size,
                              hipStream_t stream) {
  (void)in_sizes; (void)out_size; (void)ws_size;
  if (n_in < 94) return;

  const float* x_in = (const float*)d_in[0];
  int pi = 1;
  auto nxt = [&]() { return (const float*)d_in[pi++]; };

  // params flattened in JAX tree order (sorted dict keys, lists in order)
  const float* head_b  = nxt();
  const float* head_w  = nxt();
  const float* patch_b = nxt();
  const float* patch_w = nxt();

  struct MambaP { const float *A_log,*D,*conv_b,*conv_w,*dt_bias,*in_proj_w,*norm_w,*out_proj_w; };
  struct BlockP { MambaP m; const float *mlp_b1,*mlp_b2,*mlp_w1,*mlp_w2,*n2b,*n2w,*nb,*nw; };
  struct MergeP { const float *nb,*nw,*red; };
  BlockP blocks[8]; MergeP merges[3];

  int dims[4] = {96,192,384,768};
  int depths[4] = {1,1,2,1};
  int bi = 0;
  for (int s = 0; s < 4; ++s) {
    for (int d = 0; d < depths[s]; ++d) {
      BlockP& B = blocks[bi++];
      B.m.A_log = nxt(); B.m.D = nxt(); B.m.conv_b = nxt(); B.m.conv_w = nxt();
      B.m.dt_bias = nxt(); B.m.in_proj_w = nxt(); B.m.norm_w = nxt(); B.m.out_proj_w = nxt();
      B.mlp_b1 = nxt(); B.mlp_b2 = nxt(); B.mlp_w1 = nxt(); B.mlp_w2 = nxt();
      B.n2b = nxt(); B.n2w = nxt(); B.nb = nxt(); B.nw = nxt();
    }
    if (s < 3) { merges[s].nb = nxt(); merges[s].nw = nxt(); merges[s].red = nxt(); }
  }

  const int Bn = 2;
  float* wsf = (float*)d_ws;
  const size_t FBCAP = (size_t)1 << 20;          // 1M floats per feature buffer
  float* FB0 = wsf;
  float* FB1 = wsf + FBCAP;
  const size_t TRBASE = 2*FBCAP;
  size_t troff = TRBASE;
  auto talloc = [&](size_t n){ float* p = wsf + troff; troff += n; return p; };
  auto g256 = [](long n){ return (unsigned)((n + TPB - 1) / TPB); };
  auto gemm = [&](const float* A, const float* W, const float* bias, const float* res,
                  float* C, long M, int N, int K, int act){
    dim3 g((N + 63)/64, (unsigned)((M + 127)/128));
    k_gemm<<<g, 512, 0, stream>>>(A, W, bias, res, C, M, N, K, act);
  };

  // ---- patch embed ----
  float* xcur = FB0;
  k_patch<<<g256((long)Bn*64*64*96), TPB, 0, stream>>>(x_in, patch_w, patch_b, xcur, Bn);

  int S = 64;
  bi = 0;
  for (int st = 0; st < 4; ++st) {
    int d = dims[st];
    for (int dep = 0; dep < depths[st]; ++dep) {
      const BlockP& P = blocks[bi++];
      int di = 2*d, H = di/64, Dc = di + 128, dip = 2*di + 128 + H;
      long L = (long)S*S, rows = (long)Bn*S*S;
      int nseq = 4*Bn;
      long rows4 = (long)nseq*L;
      int Cn = (int)(L/64);

      troff = TRBASE;
      float* t_ln = talloc(rows*d);
      float* seqs = talloc(rows4*d);
      float* zx   = talloc(rows4*dip);
      float* dtb  = talloc(rows4*H);
      float* xbc  = talloc(rows4*Dc);
      float* Yb   = talloc(rows4*di);
      float* stt  = talloc(((size_t)nseq*Cn*H) << 12);
      float* cA   = talloc((size_t)nseq*Cn*H*64);
      float* mo   = talloc(rows4*d);
      float* tmlp = talloc(rows*4*(size_t)d);

      // Mamba branch
      k_ln<<<(unsigned)rows, 256, 0, stream>>>(xcur, P.nw, P.nb, t_ln, rows, d);
      k_gather<<<g256(rows4*d), TPB, 0, stream>>>(t_ln, seqs, Bn, S, d);
      gemm(seqs, P.m.in_proj_w, nullptr, nullptr, zx, rows4, dip, d, 0);
      k_dt<<<g256(rows4*H), TPB, 0, stream>>>(zx, P.m.dt_bias, dtb, rows4, dip, di, Dc, H);
      k_conv<<<g256((long)nseq*L*Dc), TPB, 0, stream>>>(zx, P.m.conv_w, P.m.conv_b, xbc,
                                                        nseq, (int)L, Dc, dip, di);
      k_ssd1<<<nseq*Cn*H, 512, 0, stream>>>(xbc, dtb, P.m.A_log, Yb, stt, cA,
                                            (int)L, Cn, H, Dc, di);
      k_scan<<<g256(((long)nseq*H) << 12), TPB, 0, stream>>>(stt, cA, nseq, Cn, H);
      k_ssd2<<<nseq*Cn*H, 512, 0, stream>>>(xbc, stt, cA, P.m.D, Yb,
                                            (int)L, Cn, H, Dc, di);
      k_gate_rms<<<(unsigned)rows4, 256, 0, stream>>>(Yb, zx, P.m.norm_w, rows4, di, dip);
      gemm(Yb, P.m.out_proj_w, nullptr, nullptr, mo, rows4, d, di, 0);
      k_combine<<<g256(rows*d), TPB, 0, stream>>>(xcur, mo, Bn, S, d);

      // MLP branch
      k_ln<<<(unsigned)rows, 256, 0, stream>>>(xcur, P.n2w, P.n2b, t_ln, rows, d);
      gemm(t_ln, P.mlp_w1, P.mlp_b1, nullptr, tmlp, rows, 4*d, d, 1);
      gemm(tmlp, P.mlp_w2, P.mlp_b2, xcur, xcur, rows, d, 4*d, 0);
    }
    if (st < 3) {
      const MergeP& M = merges[st];
      int d = dims[st], S2 = S/2, C4 = 4*d;
      long rowsm = (long)Bn*S2*S2;
      troff = TRBASE;
      float* t4  = talloc(rowsm*C4);
      float* t4n = talloc(rowsm*C4);
      float* xnext = (xcur == FB0) ? FB1 : FB0;
      k_merge_gather<<<g256(rowsm*C4), TPB, 0, stream>>>(xcur, t4, Bn, S, d);
      k_ln<<<(unsigned)rowsm, 256, 0, stream>>>(t4, M.nw, M.nb, t4n, rowsm, C4);
      gemm(t4n, M.red, nullptr, nullptr, xnext, rowsm, 2*d, C4, 0);
      xcur = xnext;
      S = S2;
    }
  }

  // ---- head ----
  troff = TRBASE;
  float* pooled = talloc((size_t)Bn*768);
  k_meanpool<<<g256((long)Bn*768), TPB, 0, stream>>>(xcur, pooled, Bn, S, 768);
  gemm(pooled, head_w, head_b, nullptr, (float*)d_out, Bn, 1000, 768, 0);
}
